// model_19018115186758
// MI455X (gfx1250) — matrix operand order repaired and checked by reading, not compiled, not run on hardware
//
#include <hip/hip_runtime.h>
#include <hip/hip_bf16.h>
#include <math.h>

// Problem dims
#define T_STEPS 512
#define BATCH   64
#define IDIM    256
#define HDIM    1024
#define ODIM    256
#define KDIM    (IDIM + HDIM)   // 1280
#define NB      32              // persistent blocks
#define NCOL    (HDIM / NB)     // 32 H-columns per block
#define CHUNK_K 128             // K-values per LDS chunk (4 k-tiles of 32)
#define NCHUNK  (KDIM / CHUNK_K) // 10

typedef __attribute__((ext_vector_type(16))) __bf16 v16bf;
typedef __attribute__((ext_vector_type(8)))  float  v8f;
typedef __attribute__((ext_vector_type(4)))  unsigned uint4v;

union ABFrag { v16bf v; uint4v q[2]; };

// ---- async global -> LDS (CDNA5, tracked by ASYNCcnt) --------------------
__device__ __forceinline__ void async_g2l_b128(unsigned lds_off, const void* gaddr) {
  asm volatile("global_load_async_to_lds_b128 %0, %1, off"
               :: "v"(lds_off), "v"(gaddr) : "memory");
}
__device__ __forceinline__ void wait_async_le4() {
#if __has_builtin(__builtin_amdgcn_s_wait_asynccnt)
  __builtin_amdgcn_s_wait_asynccnt(4);
#else
  asm volatile("s_wait_asynccnt 0x4" ::: "memory");
#endif
}
__device__ __forceinline__ void wait_async_0() {
#if __has_builtin(__builtin_amdgcn_s_wait_asynccnt)
  __builtin_amdgcn_s_wait_asynccnt(0);
#else
  asm volatile("s_wait_asynccnt 0x0" ::: "memory");
#endif
}

// ---- WMMA fragment loads -------------------------------------------------
// A-matrix 16x32 bf16 (ISA 7.12.2) sourced from an LDS chunk (ld = CHUNK_K):
// lane l<16 row M=l holds K=[k0..k0+7] in v0..3, K=[k0+16..k0+23] in v4..7;
// lanes 16..31 same rows, +8 K offset.
__device__ __forceinline__ v16bf lds_afrag(const __bf16* buf, int mbase, int k0,
                                           int lane) {
  int m   = mbase + (lane & 15);
  int sel = (lane >> 4) & 1;
  const __bf16* p = buf + m * CHUNK_K + (k0 + sel * 8);
  ABFrag u;
  u.q[0] = *(const uint4v*)p;        // ds_load_b128
  u.q[1] = *(const uint4v*)(p + 16); // ds_load_b128
  return u.v;
}

// A-fragment straight from global (used for the small output GEMM)
__device__ __forceinline__ v16bf load_afrag(const __bf16* __restrict__ src,
                                            int ld, int mbase, int k0, int lane) {
  int m   = mbase + (lane & 15);
  int sel = (lane >> 4) & 1;
  const __bf16* p = src + (size_t)m * ld + (k0 + sel * 8);
  ABFrag u;
  u.q[0] = *(const uint4v*)p;
  u.q[1] = *(const uint4v*)(p + 16);
  return u.v;
}

// B-matrix 32x16 bf16: lanes 0-15 hold K=0..7,16..23 at N=lane,
// lanes 16-31 hold K=8..15,24..31 at N=lane-16 (matches A). Weights stored
// K-major so this is two 16-byte b128 loads per lane.
__device__ __forceinline__ v16bf load_bfrag(const __bf16* __restrict__ wT,
                                            int ldk, int nbase, int k0, int lane) {
  int n   = nbase + (lane & 15);
  int sel = (lane >> 4) & 1;
  const __bf16* p = wT + (size_t)n * ldk + (k0 + sel * 8);
  ABFrag u;
  u.q[0] = *(const uint4v*)p;
  u.q[1] = *(const uint4v*)(p + 16);
  return u.v;
}

__device__ __forceinline__ v8f wmma_bf16(v16bf a, v16bf b, v8f c) {
  return __builtin_amdgcn_wmma_f32_16x16x32_bf16(false, a, false, b,
                                                 (short)0, c, false, false);
}

// Issue async copy of one 64 x CHUNK_K bf16 chunk (16 KB) into LDS.
// 1024 x 16B segments; each of the 256 threads issues 4 async B128 copies.
__device__ __forceinline__ void issue_chunk(const __bf16* __restrict__ src, int ld,
                                            int kbase, __bf16* dstbuf, int tid) {
#pragma unroll
  for (int j = 0; j < 4; ++j) {
    int s   = tid + 256 * j;
    int row = s >> 4;          // 0..63
    int seg = s & 15;          // 16 segments x 8 bf16 = 128 K-values
    const __bf16* gp = src + (size_t)row * ld + (kbase + seg * 8);
    unsigned loff = (unsigned)(size_t)(dstbuf + row * CHUNK_K + seg * 8);
    async_g2l_b128(loff, gp);
  }
}

// ---- device-wide generation barrier (persistent grid) --------------------
__device__ __forceinline__ void global_barrier(unsigned* bar) {
  __syncthreads();
  if (threadIdx.x == 0) {
    __threadfence();
    unsigned g = __hip_atomic_load(&bar[1], __ATOMIC_RELAXED, __HIP_MEMORY_SCOPE_AGENT);
    unsigned arrived = __hip_atomic_fetch_add(&bar[0], 1u, __ATOMIC_ACQ_REL,
                                              __HIP_MEMORY_SCOPE_AGENT);
    if (arrived == NB - 1) {
      __hip_atomic_store(&bar[0], 0u, __ATOMIC_RELAXED, __HIP_MEMORY_SCOPE_AGENT);
      __hip_atomic_fetch_add(&bar[1], 1u, __ATOMIC_RELEASE, __HIP_MEMORY_SCOPE_AGENT);
    } else {
      while (__hip_atomic_load(&bar[1], __ATOMIC_ACQUIRE,
                               __HIP_MEMORY_SCOPE_AGENT) == g)
        __builtin_amdgcn_s_sleep(1);
    }
    __threadfence();
  }
  __syncthreads();
}

// ---- prep kernels --------------------------------------------------------
__global__ void k_prep_zero(unsigned* bar, __bf16* a0) {
  size_t i = (size_t)blockIdx.x * blockDim.x + threadIdx.x;
  if (i == 0) { bar[0] = 0u; bar[1] = 0u; }
  size_t n = (size_t)BATCH * HDIM;
  for (size_t j = i; j < n; j += (size_t)gridDim.x * blockDim.x)
    a0[j] = (__bf16)0.0f;
}

__global__ void k_cvt_bf16(const float* __restrict__ src, __bf16* __restrict__ dst,
                           size_t n) {
  size_t i = (size_t)blockIdx.x * blockDim.x + threadIdx.x;
  for (size_t j = i; j < n; j += (size_t)gridDim.x * blockDim.x)
    dst[j] = (__bf16)src[j];
}

// dst[n*K + k] = (bf16) src[k*N + n]   (weights (K,N) row-major -> K-major)
__global__ void k_transpose_bf16(const float* __restrict__ src,
                                 __bf16* __restrict__ dst, int K, int N) {
  size_t total = (size_t)K * N;
  size_t i = (size_t)blockIdx.x * blockDim.x + threadIdx.x;
  for (size_t j = i; j < total; j += (size_t)gridDim.x * blockDim.x) {
    int n = (int)(j / (size_t)K);
    int k = (int)(j % (size_t)K);
    dst[j] = (__bf16)src[(size_t)k * N + n];
  }
}

// ---- persistent RNN kernel ----------------------------------------------
__global__ __launch_bounds__(256) void k_rnn(
    const float* __restrict__ bi, const float* __restrict__ b1f,
    const float* __restrict__ b2f, const float* __restrict__ bof,
    const __bf16* __restrict__ x_bf,
    __bf16* __restrict__ a_bf0, __bf16* __restrict__ a_bf1,
    __bf16* __restrict__ h_bf0, __bf16* __restrict__ h_bf1,
    const __bf16* __restrict__ w1t, const __bf16* __restrict__ w2t,
    const __bf16* __restrict__ wit, const __bf16* __restrict__ wot,
    float* __restrict__ out, float* __restrict__ h_final, unsigned* bar) {
  __shared__ __bf16 shA[2][64 * CHUNK_K];       // 2 x 16 KB double buffer

  const int tid  = threadIdx.x;
  const int lane = tid & 31;
  const int w    = tid >> 5;        // wave 0..7
  const int mt   = w & 3;           // M tile (batch rows 16*mt..)
  const int nt   = w >> 2;          // N tile within block's 32-col slice
  const int blk  = blockIdx.x;
  const int sel  = (lane >> 4) & 1;
  const int mbase = mt * 16;
  const int nbase = blk * NCOL + nt * 16;       // global H column base
  const int ngl   = nbase + (lane & 15);        // this lane's H column

  const float bb1 = b1f[ngl];
  const float bb2 = b2f[ngl];
  const float bbi = bi[ngl];

  // output-GEMM tile assignment (waves 0,1 only)
  const int mt2 = blk & 3;
  const int nt2 = (blk >> 2) * 2 + w;           // 0..15 when w<2
  float bbo = 0.0f;
  if (w < 2) bbo = bof[nt2 * 16 + (lane & 15)];

  float ho[8];                                   // h state, lives in VGPRs
#pragma unroll
  for (int r = 0; r < 8; ++r) ho[r] = 0.0f;

  for (int t = 0; t < T_STEPS; ++t) {
    const int p = t & 1;
    const __bf16* a_rd = p ? a_bf1 : a_bf0;      // tanh(h_t) broadcast, read
    __bf16* a_wr = p ? a_bf0 : a_bf1;            // tanh(h_{t+1}), write
    __bf16* h_wr = p ? h_bf1 : h_bf0;            // h_{t+1} bf16 for out GEMM
    const __bf16* h_rd = h_wr;

    // ---- Phase A: gates + input projection + state update ----
    v8f acc1 = {}; v8f acc2 = {}; v8f accw = {};
    const __bf16* xt = x_bf + (size_t)t * (BATCH * IDIM);

    issue_chunk(xt, IDIM, 0, shA[0], tid);       // prologue: chunk 0 (x part)
    for (int c = 0; c < NCHUNK; ++c) {
      if (c < NCHUNK - 1) {
        const int cn = c + 1;
        const __bf16* src = (cn < 2) ? xt : a_rd;
        const int ld  = (cn < 2) ? IDIM : HDIM;
        const int kb  = (cn < 2) ? cn * CHUNK_K : cn * CHUNK_K - IDIM;
        issue_chunk(src, ld, kb, shA[cn & 1], tid);
        wait_async_le4();                        // chunk c landed, c+1 in flight
      } else {
        wait_async_0();
      }
      __syncthreads();                           // publish chunk c to all waves

      const __bf16* buf = shA[c & 1];
#pragma unroll
      for (int ktl = 0; ktl < 4; ++ktl) {
        const int kg = c * CHUNK_K + ktl * 32;   // global K index
        v16bf a   = lds_afrag(buf, mbase, ktl * 32, lane);
        v16bf bw1 = load_bfrag(w1t, KDIM, nbase, kg, lane);
        v16bf bw2 = load_bfrag(w2t, KDIM, nbase, kg, lane);
        acc1 = wmma_bf16(a, bw1, acc1);
        acc2 = wmma_bf16(a, bw2, acc2);
        if (c < 2) {                             // x part also feeds Wi proj
          v16bf bwi = load_bfrag(wit, IDIM, nbase, kg, lane);
          accw = wmma_bf16(a, bwi, accw);
        }
      }
      __syncthreads();                           // chunk buffer reusable
    }

#pragma unroll
    for (int r = 0; r < 8; ++r) {
      const int m = mbase + sel * 8 + r;         // C/D layout: M = r + 8*sel
      float g1 = tanhf(acc1[r] + bb1);
      float g2 = tanhf(acc2[r] + bb2);
      float pre = (g1 + g2) * ho[r] + accw[r] + bbi;   // g1*h + g2*h
      float hn = 1.0f / (1.0f + __expf(-pre));
      ho[r] = hn;
      a_wr[(size_t)m * HDIM + ngl] = (__bf16)tanhf(hn);
      h_wr[(size_t)m * HDIM + ngl] = (__bf16)hn;
    }

    global_barrier(bar);

    // ---- Phase B: out_t = tanh(h_{t+1} @ Wo + bo), waves 0,1 ----
    if (w < 2) {
      v8f acc = {};
#pragma unroll 4
      for (int kt = 0; kt < 32; ++kt) {
        v16bf a  = load_afrag(h_rd, HDIM, mt2 * 16, kt * 32, lane);
        v16bf bw = load_bfrag(wot, HDIM, nt2 * 16, kt * 32, lane);
        acc = wmma_bf16(a, bw, acc);
      }
      float* op = out + (size_t)t * (BATCH * ODIM);
      const int n = nt2 * 16 + (lane & 15);
#pragma unroll
      for (int r = 0; r < 8; ++r) {
        const int m = mt2 * 16 + sel * 8 + r;
        op[(size_t)m * ODIM + n] = tanhf(acc[r] + bbo);
      }
    }
  }

  // final hidden state h1 (fp32, second output)
#pragma unroll
  for (int r = 0; r < 8; ++r) {
    const int m = mbase + sel * 8 + r;
    h_final[(size_t)m * HDIM + ngl] = ho[r];
  }
}

// ---- host launcher -------------------------------------------------------
extern "C" void kernel_launch(void* const* d_in, const int* in_sizes, int n_in,
                              void* d_out, int out_size, void* d_ws, size_t ws_size,
                              hipStream_t stream) {
  (void)in_sizes; (void)n_in; (void)out_size; (void)ws_size;
  const float* x  = (const float*)d_in[0];
  const float* Wi = (const float*)d_in[1];
  const float* bi = (const float*)d_in[2];
  const float* W1 = (const float*)d_in[3];
  const float* b1 = (const float*)d_in[4];
  const float* W2 = (const float*)d_in[5];
  const float* b2 = (const float*)d_in[6];
  const float* Wo = (const float*)d_in[7];
  const float* bo = (const float*)d_in[8];

  float* out     = (float*)d_out;
  float* h_final = out + (size_t)T_STEPS * BATCH * ODIM;

  // workspace layout (bf16 unless noted)
  char* ws = (char*)d_ws;
  size_t off = 0;
  unsigned* bar = (unsigned*)(ws + off); off += 256;
  __bf16* x_bf  = (__bf16*)(ws + off); off += (size_t)T_STEPS * BATCH * IDIM * 2;
  __bf16* a0    = (__bf16*)(ws + off); off += (size_t)BATCH * HDIM * 2;
  __bf16* a1    = (__bf16*)(ws + off); off += (size_t)BATCH * HDIM * 2;
  __bf16* hb0   = (__bf16*)(ws + off); off += (size_t)BATCH * HDIM * 2;
  __bf16* hb1   = (__bf16*)(ws + off); off += (size_t)BATCH * HDIM * 2;
  __bf16* w1t   = (__bf16*)(ws + off); off += (size_t)KDIM * HDIM * 2;
  __bf16* w2t   = (__bf16*)(ws + off); off += (size_t)KDIM * HDIM * 2;
  __bf16* wit   = (__bf16*)(ws + off); off += (size_t)IDIM * HDIM * 2;
  __bf16* wot   = (__bf16*)(ws + off); off += (size_t)HDIM * ODIM * 2;

  k_prep_zero<<<64, 256, 0, stream>>>(bar, a0);
  k_cvt_bf16<<<2048, 256, 0, stream>>>(x, x_bf, (size_t)T_STEPS * BATCH * IDIM);
  k_transpose_bf16<<<1024, 256, 0, stream>>>(W1, w1t, KDIM, HDIM);
  k_transpose_bf16<<<1024, 256, 0, stream>>>(W2, w2t, KDIM, HDIM);
  k_transpose_bf16<<<256, 256, 0, stream>>>(Wi, wit, IDIM, HDIM);
  k_transpose_bf16<<<256, 256, 0, stream>>>(Wo, wot, HDIM, ODIM);

  k_rnn<<<NB, 256, 0, stream>>>(bi, b1, b2, bo, x_bf, a0, a1, hb0, hb1,
                                w1t, w2t, wit, wot, out, h_final, bar);
}
